// GraphAttentionInfluence_14465449853681
// MI455X (gfx1250) — matrix operand order repaired and checked by reading, not compiled, not run on hardware
//
#include <hip/hip_runtime.h>
#include <hip/hip_bf16.h>
#include <math.h>

typedef __attribute__((ext_vector_type(16))) __bf16 v16bf;
typedef __attribute__((ext_vector_type(8)))  __bf16 v8bf;
typedef __attribute__((ext_vector_type(8)))  float  v8f;

#define NEG_SLOPE 0.2f

// CDNA5 async global->LDS copy path (ASYNCcnt), guarded so compile never fails.
#if defined(__has_builtin)
# if __has_builtin(__builtin_amdgcn_global_load_async_to_lds_b128) && \
     __has_builtin(__builtin_amdgcn_s_wait_asynccnt)
#  define USE_ASYNC_LDS 1
# endif
#endif
#ifndef USE_ASYNC_LDS
# define USE_ASYNC_LDS 0
#endif

#if USE_ASYNC_LDS
// builtin prototype (from clang diagnostic): v4i addrspace(1)* src, v4i addrspace(3)* dst, imm, imm
typedef int v4i_ __attribute__((vector_size(16)));
typedef __attribute__((address_space(1))) v4i_* as1_v4i;
typedef __attribute__((address_space(3))) v4i_* as3_v4i;
#endif

// ---------- helpers ----------
__device__ __forceinline__ unsigned short f2bf(float f) {
  unsigned u = __float_as_uint(f);
  u += 0x7FFFu + ((u >> 16) & 1u);          // round-to-nearest-even
  return (unsigned short)(u >> 16);
}
__device__ __forceinline__ float bf2f(unsigned short b) {
  return __uint_as_float(((unsigned)b) << 16);
}
// monotonic uint key so unsigned atomicMax == float max (init 0 == -inf)
__device__ __forceinline__ unsigned fkey(float f) {
  unsigned u = __float_as_uint(f);
  return (u & 0x80000000u) ? ~u : (u | 0x80000000u);
}
__device__ __forceinline__ float funkey(unsigned k) {
  unsigned u = (k & 0x80000000u) ? (k & 0x7FFFFFFFu) : ~k;
  return __uint_as_float(u);
}
__device__ __forceinline__ v16bf cat16(v8bf lo, v8bf hi) {
  return __builtin_shufflevector(lo, hi, 0,1,2,3,4,5,6,7,8,9,10,11,12,13,14,15);
}

// ---------- elementwise prep ----------
__global__ void k_zero(unsigned* __restrict__ p, long long n) {
  long long i = (long long)blockIdx.x * blockDim.x + threadIdx.x;
  long long st = (long long)gridDim.x * blockDim.x;
  for (; i < n; i += st) p[i] = 0u;
}

__global__ void k_cast_bf16(const float* __restrict__ in, unsigned short* __restrict__ out,
                            long long n) {
  long long i = (long long)blockIdx.x * blockDim.x + threadIdx.x;
  long long st = (long long)gridDim.x * blockDim.x;
  for (; i < n; i += st) out[i] = f2bf(in[i]);
}

// W [K, Nout] row-major f32  ->  WT [Nout, K] row-major bf16 (column-major B)
__global__ void k_transpose_bf16(const float* __restrict__ W, unsigned short* __restrict__ WT,
                                 int K, int Nout) {
  long long n = (long long)K * Nout;
  long long i = (long long)blockIdx.x * blockDim.x + threadIdx.x;
  long long st = (long long)gridDim.x * blockDim.x;
  for (; i < n; i += st) {
    int k = (int)(i / Nout), c = (int)(i % Nout);
    WT[(long long)c * K + k] = f2bf(W[i]);
  }
}

// ---------- WMMA GEMM with LDS-staged B ----------
// C[M,Nout] = A[M,K] * B[K,Nout]; A row-major bf16, BT=[Nout,K] row-major bf16.
// Block: 256 threads = 8 waves -> 128x64 C macro-tile. Each wave: 16x64 strip
// (4 accumulators, one A fragment reused across 4 WMMAs per K-step).
// B strip (64 cols x 64 k) double-buffered in LDS; staged with async
// global->LDS copies (ASYNCcnt) when available. LDS rows padded to 144 B so
// the 16 rows touched by a ds_load_b128 land on distinct bank groups.
// Requires: K % 64 == 0, Nout % 64 == 0 (true here: K in {384,1024}, Nout in {1024,256}).
#define BROW 72                    // 64 bf16 + 8 pad
#define BBUF (64 * BROW)           // one buffer, elems

__global__ void k_gemm_bf16_wmma(const unsigned short* __restrict__ A,
                                 const unsigned short* __restrict__ BT,
                                 unsigned short* __restrict__ C,
                                 int M, int K, int Nout) {
  __shared__ unsigned short sB[2][BBUF];

  const int bn  = (int)(blockIdx.x) % (Nout >> 6);
  const int bm  = (int)(blockIdx.x) / (Nout >> 6);
  const int colBase = bn << 6;
  const int w    = threadIdx.x >> 5;
  const int lane = threadIdx.x & 31;
  const int l15  = lane & 15;
  const int hi   = lane >> 4;              // half-wave select

  int arow = bm * 128 + w * 16 + l15;
  if (arow >= M) arow = M - 1;             // clamped duplicate; store is guarded
  // A lane layout (16-bit 16x32): elems 0..7 -> K=hi*8+0..7, elems 8..15 -> K=hi*8+16..23
  const unsigned short* ap = A + (long long)arow * K + hi * 8;

  const int nk = K >> 6;                   // 64-wide K chunks

  // stage one 64x64 K-chunk of B into sB[buf]
  auto stage = [&](int kc, int buf) {
    const int t = threadIdx.x;
    #pragma unroll
    for (int it = 0; it < 2; it++) {
      int item = t + it * 256;             // 512 x 16B items
      int r  = item >> 3;                  // B column (LDS row), 0..63
      int c8 = item & 7;                   // 16B chunk within 64 k-values
      const unsigned short* g = BT + (long long)(colBase + r) * K + (kc << 6) + (c8 << 3);
      unsigned short* l = &sB[buf][r * BROW + (c8 << 3)];
#if USE_ASYNC_LDS
      __builtin_amdgcn_global_load_async_to_lds_b128((as1_v4i)(void*)g, (as3_v4i)(void*)l, 0, 0);
#else
      *(uint4*)l = *(const uint4*)g;
#endif
    }
  };
  auto wait_stage = [&]() {
#if USE_ASYNC_LDS
    __builtin_amdgcn_s_wait_asynccnt(0);
#endif
    __syncthreads();
  };

  v8f acc[4] = {};
  stage(0, 0);
  wait_stage();

  for (int kc = 0; kc < nk; kc++) {
    int cur = kc & 1;
    if (kc + 1 < nk) stage(kc + 1, cur ^ 1);   // overlap copy with compute
    const unsigned short* lb = &sB[cur][0];
    #pragma unroll
    for (int ks = 0; ks < 64; ks += 32) {
      int kg = (kc << 6) + ks;
      __builtin_prefetch(ap + kg + 128, 0, 1);
      v16bf av = cat16(*(const v8bf*)(ap + kg), *(const v8bf*)(ap + kg + 16));
      #pragma unroll
      for (int nb = 0; nb < 4; nb++) {
        // B lane layout (16-bit 32x16): elems 0..7 -> K = hi*8 + 0..7, elems 8..15 -> K = hi*8 + 16..23
        const unsigned short* bl = lb + (nb * 16 + l15) * BROW + ks + hi * 8;
        v16bf bv = cat16(*(const v8bf*)bl, *(const v8bf*)(bl + 16));
        acc[nb] = __builtin_amdgcn_wmma_f32_16x16x32_bf16(false, av, false, bv,
                                                          (short)0, acc[nb], false, false);
      }
    }
    wait_stage();   // next buffer ready; everyone done with cur before it is rewritten
  }

  // D layout: VGPR r -> M = rbase + r + hi*8 ; N = col + (lane&15)
  int rbase = bm * 128 + w * 16 + hi * 8;
  #pragma unroll
  for (int nb = 0; nb < 4; nb++) {
    int col = colBase + nb * 16 + l15;
    #pragma unroll
    for (int r = 0; r < 8; r++) {
      int row = rbase + r;
      if (row < M) C[(long long)row * Nout + col] = f2bf(acc[nb][r]);
    }
  }
}

// ---------- attention ----------
__global__ void k_scores(const unsigned short* __restrict__ HX,
                         const float* __restrict__ asrc, const float* __restrict__ adst,
                         float* __restrict__ ss, float* __restrict__ sd,
                         int N, int H, int Cc) {
  int idx = blockIdx.x * blockDim.x + threadIdx.x;
  if (idx >= N * H) return;
  int n = idx / H, h = idx % H;
  const unsigned short* row = HX + (long long)n * H * Cc + (long long)h * Cc;
  const float* as = asrc + h * Cc;
  const float* ad = adst + h * Cc;
  float s1 = 0.f, s2 = 0.f;
  for (int c = 0; c < Cc; c++) { float v = bf2f(row[c]); s1 += v * as[c]; s2 += v * ad[c]; }
  ss[idx] = s1; sd[idx] = s2;
}

__device__ __forceinline__ void edge_sd(const int* ei, int e, int E, int& s, int& d) {
  if (e < E) { s = ei[e]; d = ei[E + e]; } else { s = d = e - E; }   // appended self-loops
}

__global__ void k_edge_max(const int* __restrict__ ei, const float* __restrict__ ss,
                           const float* __restrict__ sd, unsigned* __restrict__ mk,
                           int E, int N, int H) {
  long long tot = (long long)(E + N) * H;
  long long i = (long long)blockIdx.x * blockDim.x + threadIdx.x;
  if (i >= tot) return;
  int e = (int)(i / H), h = (int)(i % H);
  int s, d; edge_sd(ei, e, E, s, d);
  float v = ss[s * H + h] + sd[d * H + h];
  v = v > 0.f ? v : NEG_SLOPE * v;
  atomicMax(mk + d * H + h, fkey(v));
}

__global__ void k_edge_exp(const int* __restrict__ ei, const float* __restrict__ ss,
                           const float* __restrict__ sd, const unsigned* __restrict__ mk,
                           float* __restrict__ dn, float* __restrict__ ex,
                           int E, int N, int H) {
  long long tot = (long long)(E + N) * H;
  long long i = (long long)blockIdx.x * blockDim.x + threadIdx.x;
  if (i >= tot) return;
  int e = (int)(i / H), h = (int)(i % H);
  int s, d; edge_sd(ei, e, E, s, d);
  float v = ss[s * H + h] + sd[d * H + h];
  v = v > 0.f ? v : NEG_SLOPE * v;
  float m = funkey(mk[d * H + h]);
  float z = __expf(v - m);
  ex[i] = z;
  atomicAdd(dn + d * H + h, z);
}

// wave per edge: agg[dst, :] += h[src, :] * alpha[e, head(c)] ; Cc = 1<<cshift
__global__ void k_edge_agg(const int* __restrict__ ei, const unsigned short* __restrict__ HX,
                           const float* __restrict__ ex, const float* __restrict__ dn,
                           float* __restrict__ agg, int E, int N, int H, int cshift) {
  int F = H << cshift;
  long long wid = ((long long)blockIdx.x * blockDim.x + threadIdx.x) >> 5;
  int lane = threadIdx.x & 31;
  if (wid >= (long long)(E + N)) return;
  int e = (int)wid;
  int s, d; edge_sd(ei, e, E, s, d);
  float al[4];
  #pragma unroll
  for (int h = 0; h < 4; h++)
    if (h < H) al[h] = ex[(long long)e * H + h] / (dn[d * H + h] + 1e-16f);
  const unsigned short* hs = HX + (long long)s * F;
  float* ad = agg + (long long)d * F;
  for (int c = lane; c < F; c += 32)
    atomicAdd(ad + c, bf2f(hs[c]) * al[c >> cshift]);
}

// ---------- epilogues ----------
__global__ void k_bias_elu_bf16(const float* __restrict__ agg, const float* __restrict__ b,
                                unsigned short* __restrict__ out, long long n, int F) {
  long long i = (long long)blockIdx.x * blockDim.x + threadIdx.x;
  long long st = (long long)gridDim.x * blockDim.x;
  for (; i < n; i += st) {
    float v = agg[i] + b[(int)(i % F)];
    v = v > 0.f ? v : (__expf(v) - 1.f);
    out[i] = f2bf(v);
  }
}

__global__ void k_bias_elu_f32(const float* __restrict__ agg, const float* __restrict__ b,
                               float* __restrict__ out, long long n, int F) {
  long long i = (long long)blockIdx.x * blockDim.x + threadIdx.x;
  long long st = (long long)gridDim.x * blockDim.x;
  for (; i < n; i += st) {
    float v = agg[i] + b[(int)(i % F)];
    out[i] = v > 0.f ? v : (__expf(v) - 1.f);
  }
}

// ---------- readout head ----------
__global__ void k_graph_mean(const float* __restrict__ h, float* __restrict__ gf, int N, int C) {
  int c = threadIdx.x;
  float s = 0.f;
  for (int r = blockIdx.x; r < N; r += gridDim.x) s += h[(long long)r * C + c];
  atomicAdd(gf + c, s / (float)N);
}

__global__ void k_infl(const float* __restrict__ h, const float* __restrict__ Wp1,
                       const float* __restrict__ bp1, const float* __restrict__ Wp2,
                       const float* __restrict__ bp2, float* __restrict__ out,
                       int N, int C, int Hh) {
  __shared__ float row[256];
  __shared__ float red[128];
  int n = blockIdx.x;
  int t = threadIdx.x;
  for (int i = t; i < C; i += blockDim.x) row[i] = h[(long long)n * C + i];
  __syncthreads();
  float acc = bp1[t];
  for (int k = 0; k < C; k++) acc += row[k] * Wp1[k * Hh + t];
  float hid = acc > 0.f ? acc : 0.f;
  red[t] = hid * Wp2[t];
  __syncthreads();
  for (int sft = 64; sft > 0; sft >>= 1) { if (t < sft) red[t] += red[t + sft]; __syncthreads(); }
  if (t == 0) out[n] = 1.f / (1.f + __expf(-(red[0] + bp2[0])));
}

// ---------- host ----------
static inline int cdiv_i(long long a, int b) { return (int)((a + b - 1) / b); }

extern "C" void kernel_launch(void* const* d_in, const int* in_sizes, int n_in,
                              void* d_out, int out_size, void* d_ws, size_t ws_size,
                              hipStream_t stream) {
  (void)n_in; (void)out_size; (void)ws_size;
  const float* x      = (const float*)d_in[0];
  const int*   ei     = (const int*)  d_in[1];
  const float* W1     = (const float*)d_in[2];
  const float* a_src1 = (const float*)d_in[3];
  const float* a_dst1 = (const float*)d_in[4];
  const float* b1     = (const float*)d_in[5];
  const float* W2     = (const float*)d_in[6];
  const float* a_src2 = (const float*)d_in[7];
  const float* a_dst2 = (const float*)d_in[8];
  const float* b2     = (const float*)d_in[9];
  const float* Wp1    = (const float*)d_in[10];
  const float* bp1    = (const float*)d_in[11];
  const float* Wp2    = (const float*)d_in[12];
  const float* bp2    = (const float*)d_in[13];

  const int FIN = 384, HH = 4, HD = 256, F1 = HH * HD, HMLP = 128;
  const int N  = in_sizes[0] / FIN;
  const int E  = in_sizes[1] / 2;
  const int ET = E + N;

  char* ws = (char*)d_ws;
  size_t off = 0;
  auto alloc = [&](size_t bytes) -> void* {
    void* p = (void*)(ws + off);
    off = (off + bytes + 255) & ~((size_t)255);
    return p;
  };

  unsigned short* X16   = (unsigned short*)alloc((size_t)N * FIN * 2);
  unsigned short* W1T   = (unsigned short*)alloc((size_t)F1 * FIN * 2);
  unsigned short* W2T   = (unsigned short*)alloc((size_t)HD * F1 * 2);
  unsigned short* HX16  = (unsigned short*)alloc((size_t)N * F1 * 2);
  float*          AGG1  = (float*)alloc((size_t)N * F1 * 4);
  unsigned short* ACT16 = (unsigned short*)alloc((size_t)N * F1 * 2);
  float*          S1S   = (float*)alloc((size_t)N * HH * 4);
  float*          S1D   = (float*)alloc((size_t)N * HH * 4);
  unsigned*       MK    = (unsigned*)alloc((size_t)N * HH * 4);
  float*          DN    = (float*)alloc((size_t)N * HH * 4);
  float*          EX    = (float*)alloc((size_t)ET * HH * 4);
  // aliases (sequential stream ordering makes reuse safe)
  unsigned short* HX2   = X16;    // N*HD*2 <= N*FIN*2, x consumed after GEMM1
  float*          AGG2  = AGG1;   // reused after epilogue1 consumed AGG1
  float*          EX2   = EX;

  float* hout = (float*)d_out;             // [N, 256]
  float* gf   = hout + (size_t)N * HD;     // [256]
  float* infl = gf + HD;                   // [N]

  const int TB = 256, GS = 2048;

  // ---- prep: casts / transposes ----
  k_cast_bf16<<<GS, TB, 0, stream>>>(x, X16, (long long)N * FIN);
  k_transpose_bf16<<<GS, TB, 0, stream>>>(W1, W1T, FIN, F1);
  k_transpose_bf16<<<GS, TB, 0, stream>>>(W2, W2T, F1, HD);

  // ---- layer 1: GEMM (bf16 WMMA, LDS-staged B) ----
  k_gemm_bf16_wmma<<<cdiv_i(N, 128) * (F1 / 64), TB, 0, stream>>>(X16, W1T, HX16, N, FIN, F1);

  // ---- layer 1: attention ----
  k_scores<<<cdiv_i((long long)N * HH, TB), TB, 0, stream>>>(HX16, a_src1, a_dst1, S1S, S1D, N, HH, HD);
  k_zero<<<GS, TB, 0, stream>>>(MK, (long long)N * HH);
  k_zero<<<GS, TB, 0, stream>>>((unsigned*)DN, (long long)N * HH);
  k_edge_max<<<cdiv_i((long long)ET * HH, TB), TB, 0, stream>>>(ei, S1S, S1D, MK, E, N, HH);
  k_edge_exp<<<cdiv_i((long long)ET * HH, TB), TB, 0, stream>>>(ei, S1S, S1D, MK, DN, EX, E, N, HH);
  k_zero<<<GS, TB, 0, stream>>>((unsigned*)AGG1, (long long)N * F1);
  k_edge_agg<<<cdiv_i((long long)ET * 32, TB), TB, 0, stream>>>(ei, HX16, EX, DN, AGG1, E, N, HH, 8);
  k_bias_elu_bf16<<<GS, TB, 0, stream>>>(AGG1, b1, ACT16, (long long)N * F1, F1);

  // ---- layer 2: GEMM (bf16 WMMA, LDS-staged B) ----
  k_gemm_bf16_wmma<<<cdiv_i(N, 128) * (HD / 64), TB, 0, stream>>>(ACT16, W2T, HX2, N, F1, HD);

  // ---- layer 2: attention (H=1) ----
  k_scores<<<cdiv_i((long long)N, TB), TB, 0, stream>>>(HX2, a_src2, a_dst2, S1S, S1D, N, 1, HD);
  k_zero<<<GS, TB, 0, stream>>>(MK, (long long)N);
  k_zero<<<GS, TB, 0, stream>>>((unsigned*)DN, (long long)N);
  k_edge_max<<<cdiv_i((long long)ET, TB), TB, 0, stream>>>(ei, S1S, S1D, MK, E, N, 1);
  k_edge_exp<<<cdiv_i((long long)ET, TB), TB, 0, stream>>>(ei, S1S, S1D, MK, DN, EX2, E, N, 1);
  k_zero<<<GS, TB, 0, stream>>>((unsigned*)AGG2, (long long)N * HD);
  k_edge_agg<<<cdiv_i((long long)ET * 32, TB), TB, 0, stream>>>(ei, HX2, EX2, DN, AGG2, E, N, 1, 8);
  k_bias_elu_f32<<<GS, TB, 0, stream>>>(AGG2, b2, hout, (long long)N * HD, HD);

  // ---- readout ----
  k_zero<<<1, HD, 0, stream>>>((unsigned*)gf, (long long)HD);
  k_graph_mean<<<256, HD, 0, stream>>>(hout, gf, N, HD);
  k_infl<<<N, HMLP, 0, stream>>>(hout, Wp1, bp1, Wp2, bp2, infl, N, HD, HMLP);
}
